// Attention_32701880992127
// MI455X (gfx1250) — matrix operand order repaired and checked by reading, not compiled, not run on hardware
//
#include <hip/hip_runtime.h>
#include <hip/hip_bf16.h>

// ---------------------------------------------------------------------------
// Types for CDNA5 WMMA bf16 16x16x32
// ---------------------------------------------------------------------------
typedef __attribute__((ext_vector_type(16))) __bf16 v16bf;
typedef __attribute__((ext_vector_type(8)))  float  v8f;
typedef __attribute__((ext_vector_type(4)))  unsigned int uint4v;

union FragBF {
    v16bf  v;
    uint4v u[2];
};

#define S_LEN    2048
#define D_MODEL  2048
#define HD       128
#define NH       16
#define NKV      4
#define QKV_N    3072          // 2048 q + 512 k + 512 v output features
#define RMS_EPS  1.1920929e-07f
#define SM_SCALE 0.08838834764831845f   // 1/sqrt(128)

__device__ __forceinline__ unsigned short f2bf(float f) {
    unsigned int u = __float_as_uint(f);
    unsigned int r = u + 0x7FFFu + ((u >> 16) & 1u);   // round-to-nearest-even
    return (unsigned short)(r >> 16);
}

__device__ __forceinline__ float rowmax16(float v) {
    v = fmaxf(v, __shfl_xor(v, 1, 32));
    v = fmaxf(v, __shfl_xor(v, 2, 32));
    v = fmaxf(v, __shfl_xor(v, 4, 32));
    v = fmaxf(v, __shfl_xor(v, 8, 32));
    return v;
}
__device__ __forceinline__ float rowsum16(float v) {
    v += __shfl_xor(v, 1, 32);
    v += __shfl_xor(v, 2, 32);
    v += __shfl_xor(v, 4, 32);
    v += __shfl_xor(v, 8, 32);
    return v;
}

// ---------------------------------------------------------------------------
// CDNA5 async global->LDS copy (ASYNCcnt-tracked).  Generic pointers to
// __shared__ carry the LDS byte offset in their low 32 bits (ISA 10.2);
// the instruction takes the LDS address in VDST and
// the 64-bit global address in VADDR.
// ---------------------------------------------------------------------------
__device__ __forceinline__ void async_load_b128(const void* gptr, void* lptr) {
    unsigned lds_addr = (unsigned)(unsigned long long)(size_t)lptr;
    unsigned long long gaddr = (unsigned long long)(size_t)gptr;
    asm volatile("global_load_async_to_lds_b128 %0, %1, off"
                 :: "v"(lds_addr), "v"(gaddr) : "memory");
}
__device__ __forceinline__ void wait_async_le4() {
    asm volatile("s_wait_asynccnt 4" ::: "memory");
}
__device__ __forceinline__ void wait_async_zero() {
    asm volatile("s_wait_asynccnt 0" ::: "memory");
}

// ---------------------------------------------------------------------------
// Kernel 1: fp32 -> bf16 elementwise convert
// ---------------------------------------------------------------------------
__global__ void cvt_bf16_kernel(const float* __restrict__ src,
                                unsigned short* __restrict__ dst, int n) {
    int i = blockIdx.x * blockDim.x + threadIdx.x;
    if (i < n) dst[i] = f2bf(src[i]);
}

// ---------------------------------------------------------------------------
// Kernel 2: bf16 WMMA GEMM.  C[M,N] = A[M,K] * B[N,K]^T  (C fp32)
//
// Workgroup = 256 threads (8 waves), block tile 128(M) x 128(N), K-step 32.
// A/B K-slices are staged in LDS with double-buffered async copies
// (global_load_async_to_lds_b128, s_wait_asynccnt) so the next slice's fill
// overlaps the current slice's 64 WMMAs.  Wave grid 4(M) x 2(N): each wave
// computes 32x64 = 2 A-frags x 4 B-frags -> 8 wmma per K-step.
//
// Fragment layouts per CDNA5 ISA 7.12.2:
//   A 16x32 bf16 : lane l -> row (l&15); half h: elems0-7 = K 8h..8h+7,
//                  elems8-15 = K 16+8h..23+8h
//   B 32x16 bf16 : lane l -> col (l&15); half h: same K order as A
//   C 16x16 f32  : lane l -> col (l&15); VGPR i -> row i + 8*(l>>4)
// ---------------------------------------------------------------------------
__global__ void gemm_bf16_kernel(const unsigned short* __restrict__ A,
                                 const unsigned short* __restrict__ B,
                                 float* __restrict__ C,
                                 int M, int N, int K) {
    __shared__ unsigned short As[2][128 * 32];   // [row][k] 8KB per buffer
    __shared__ unsigned short Bs[2][128 * 32];   // [col][k]

    const int tilesN = N >> 7;                   // 128-wide block columns
    const int bm0 = (blockIdx.x / tilesN) * 128;
    const int bn0 = (blockIdx.x % tilesN) * 128;

    const int tid  = threadIdx.x;
    const int wid  = tid >> 5;
    const int lane = tid & 31;
    const int lo   = lane & 15;
    const int hf   = lane >> 4;
    const int wm   = (wid >> 1) * 32;            // wave M offset in block
    const int wn   = (wid & 1) * 64;             // wave N offset in block

    const unsigned short* Ag = A + (size_t)bm0 * K;
    const unsigned short* Bg = B + (size_t)bn0 * K;

    // each thread copies 2 A-chunks + 2 B-chunks of 16B per K-slice
    const int c0r = tid >> 2,          c0k = (tid & 3) * 8;
    const int c1r = (tid + 256) >> 2,  c1k = ((tid + 256) & 3) * 8;

    v8f acc[2][4];
    #pragma unroll
    for (int s = 0; s < 2; ++s)
        #pragma unroll
        for (int t = 0; t < 4; ++t)
            acc[s][t] = (v8f){0.f,0.f,0.f,0.f,0.f,0.f,0.f,0.f};

    const int nsteps = K >> 5;

    // prologue: fill buffer 0 (4 async-load instructions per wave)
    {
        const int k0 = 0;
        async_load_b128(Ag + (size_t)c0r * K + k0 + c0k, &As[0][c0r * 32 + c0k]);
        async_load_b128(Ag + (size_t)c1r * K + k0 + c1k, &As[0][c1r * 32 + c1k]);
        async_load_b128(Bg + (size_t)c0r * K + k0 + c0k, &Bs[0][c0r * 32 + c0k]);
        async_load_b128(Bg + (size_t)c1r * K + k0 + c1k, &Bs[0][c1r * 32 + c1k]);
    }

    for (int i = 0; i < nsteps; ++i) {
        const int buf = i & 1;
        if (i + 1 < nsteps) {
            const int k0 = (i + 1) << 5;
            const int nb = buf ^ 1;
            async_load_b128(Ag + (size_t)c0r * K + k0 + c0k, &As[nb][c0r * 32 + c0k]);
            async_load_b128(Ag + (size_t)c1r * K + k0 + c1k, &As[nb][c1r * 32 + c1k]);
            async_load_b128(Bg + (size_t)c0r * K + k0 + c0k, &Bs[nb][c0r * 32 + c0k]);
            async_load_b128(Bg + (size_t)c1r * K + k0 + c1k, &Bs[nb][c1r * 32 + c1k]);
            wait_async_le4();    // oldest 4 (this buf) have landed
        } else {
            wait_async_zero();
        }
        __syncthreads();         // all waves' fills for `buf` visible

        FragBF af[2];
        #pragma unroll
        for (int s = 0; s < 2; ++s) {
            const unsigned short* ar = &As[buf][(wm + s * 16 + lo) * 32];
            af[s].u[0] = *(const uint4v*)(ar + 8 * hf);
            af[s].u[1] = *(const uint4v*)(ar + 16 + 8 * hf);
        }
        #pragma unroll
        for (int t = 0; t < 4; ++t) {
            const unsigned short* br = &Bs[buf][(wn + t * 16 + lo) * 32];
            FragBF bf;
            bf.u[0] = *(const uint4v*)(br + 8 * hf);
            bf.u[1] = *(const uint4v*)(br + 16 + 8 * hf);
            #pragma unroll
            for (int s = 0; s < 2; ++s)
                acc[s][t] = __builtin_amdgcn_wmma_f32_16x16x32_bf16(
                    false, af[s].v, false, bf.v, (short)0, acc[s][t], false, false);
        }
        __syncthreads();         // done reading `buf` before it is refilled
    }

    #pragma unroll
    for (int s = 0; s < 2; ++s)
        #pragma unroll
        for (int t = 0; t < 4; ++t) {
            const int n = bn0 + wn + t * 16 + lo;
            #pragma unroll
            for (int i = 0; i < 8; ++i) {
                const int m = bm0 + wm + s * 16 + i + 8 * hf;
                C[(size_t)m * N + n] = acc[s][t][i];
            }
        }
}

// ---------------------------------------------------------------------------
// Kernel 3: fused RMS-norm + RoPE -> bf16; also builds V^T.
// One block = 128 threads handles one (s, slot); slots 0-15=q, 16-19=k, 20-23=v.
//   q_bf16 [16][S][128], k_bf16 [4][S][128], vT_bf16 [4][128][S]
// ---------------------------------------------------------------------------
__global__ void norm_rope_kernel(const float* __restrict__ qkv,
                                 const float* __restrict__ freqs, // [S][64][2]
                                 unsigned short* __restrict__ qb,
                                 unsigned short* __restrict__ kb,
                                 unsigned short* __restrict__ vtb) {
    __shared__ float red[128];
    __shared__ float sh[128];

    const int blk  = blockIdx.x;
    const int s    = blk / 24;
    const int slot = blk % 24;
    const int d    = threadIdx.x;

    int base;
    if (slot < 16)      base = slot * HD;                 // q
    else if (slot < 20) base = 2048 + (slot - 16) * HD;   // k
    else                base = 2560 + (slot - 20) * HD;   // v

    const float val = qkv[(size_t)s * QKV_N + base + d];

    if (slot < 20) {
        red[d] = val * val;
        __syncthreads();
        for (int st = 64; st > 0; st >>= 1) {
            if (d < st) red[d] += red[d + st];
            __syncthreads();
        }
        const float inv = rsqrtf(red[0] * (1.0f / HD) + RMS_EPS);
        sh[d] = val * inv;
        __syncthreads();
        const int   p  = d >> 1;
        const float c  = freqs[(size_t)s * HD + 2 * p];
        const float sn = freqs[(size_t)s * HD + 2 * p + 1];
        float o;
        if ((d & 1) == 0) o = sh[d] * c - sh[d + 1] * sn;
        else              o = sh[d - 1] * sn + sh[d] * c;
        if (slot < 16) qb[((size_t)slot * S_LEN + s) * HD + d] = f2bf(o);
        else           kb[((size_t)(slot - 16) * S_LEN + s) * HD + d] = f2bf(o);
    } else {
        vtb[((size_t)(slot - 20) * HD + d) * S_LEN + s] = f2bf(val);
    }
}

// ---------------------------------------------------------------------------
// Kernel 4: flash attention with online softmax + sink gating.
// One wave per (head, 16-row Q tile).  Keys processed 32 at a time.
//   S-tile:  4x wmma(q, k)        P relaid via LDS into A-fragment
//   O-tile:  8x wmma(P, vT)
// ---------------------------------------------------------------------------
__global__ void flash_attn_kernel(const unsigned short* __restrict__ qb,
                                  const unsigned short* __restrict__ kb,
                                  const unsigned short* __restrict__ vtb,
                                  const float* __restrict__ sinks,
                                  unsigned short* __restrict__ att) {
    __shared__ unsigned short Pl[16 * 32];

    const int h    = blockIdx.y;
    const int m0   = blockIdx.x * 16;
    const int hkv  = h >> 2;               // 4:1 GQA
    const int lane = threadIdx.x & 31;
    const int lo   = lane & 15;
    const int hf   = lane >> 4;

    FragBF qf[4];
    {
        const unsigned short* qrow = qb + ((size_t)h * S_LEN + (m0 + lo)) * HD;
        #pragma unroll
        for (int c = 0; c < 4; ++c) {
            qf[c].u[0] = *(const uint4v*)(qrow + c * 32 + 8 * hf);
            qf[c].u[1] = *(const uint4v*)(qrow + c * 32 + 16 + 8 * hf);
        }
    }

    v8f Ot[8];
    #pragma unroll
    for (int t = 0; t < 8; ++t) Ot[t] = (v8f){0.f,0.f,0.f,0.f,0.f,0.f,0.f,0.f};
    float rmax[8], rsum[8];
    #pragma unroll
    for (int i = 0; i < 8; ++i) { rmax[i] = -1e30f; rsum[i] = 0.f; }

    const int jend = m0 + 16;
    for (int j0 = 0; j0 < jend; j0 += 32) {
        v8f s0 = (v8f){0.f,0.f,0.f,0.f,0.f,0.f,0.f,0.f};
        v8f s1 = s0;
        const unsigned short* k0row = kb + ((size_t)hkv * S_LEN + (j0 + lo)) * HD;
        const unsigned short* k1row = kb + ((size_t)hkv * S_LEN + (j0 + 16 + lo)) * HD;
        #pragma unroll
        for (int c = 0; c < 4; ++c) {
            FragBF kf;
            kf.u[0] = *(const uint4v*)(k0row + c * 32 + 8 * hf);
            kf.u[1] = *(const uint4v*)(k0row + c * 32 + 16 + 8 * hf);
            s0 = __builtin_amdgcn_wmma_f32_16x16x32_bf16(
                false, qf[c].v, false, kf.v, (short)0, s0, false, false);
            kf.u[0] = *(const uint4v*)(k1row + c * 32 + 8 * hf);
            kf.u[1] = *(const uint4v*)(k1row + c * 32 + 16 + 8 * hf);
            s1 = __builtin_amdgcn_wmma_f32_16x16x32_bf16(
                false, qf[c].v, false, kf.v, (short)0, s1, false, false);
        }

        __syncthreads();   // single-wave block: protects LDS reuse across iters

        float alpha[8];
        #pragma unroll
        for (int i = 0; i < 8; ++i) {
            const int gm   = m0 + i + 8 * hf;
            const int col0 = j0 + lo;
            const int col1 = j0 + 16 + lo;
            float f0 = s0[i] * SM_SCALE;
            float f1 = s1[i] * SM_SCALE;
            if (col0 > gm) f0 = -1e30f;
            if (col1 > gm) f1 = -1e30f;
            const float bm   = rowmax16(fmaxf(f0, f1));
            const float newm = fmaxf(rmax[i], bm);
            alpha[i] = __expf(rmax[i] - newm);
            const float p0 = __expf(f0 - newm);
            const float p1 = __expf(f1 - newm);
            rsum[i] = rsum[i] * alpha[i] + rowsum16(p0 + p1);
            rmax[i] = newm;
            Pl[(i + 8 * hf) * 32 + lo]      = f2bf(p0);
            Pl[(i + 8 * hf) * 32 + 16 + lo] = f2bf(p1);
        }
        __syncthreads();

        FragBF pf;
        pf.u[0] = *(const uint4v*)(&Pl[lo * 32 + 8 * hf]);
        pf.u[1] = *(const uint4v*)(&Pl[lo * 32 + 16 + 8 * hf]);

        #pragma unroll
        for (int t = 0; t < 8; ++t) {
            #pragma unroll
            for (int i = 0; i < 8; ++i) Ot[t][i] *= alpha[i];
            const unsigned short* vrow =
                vtb + ((size_t)hkv * HD + t * 16 + lo) * S_LEN + j0 + 8 * hf;
            FragBF vf;
            vf.u[0] = *(const uint4v*)(vrow);
            vf.u[1] = *(const uint4v*)(vrow + 16);
            Ot[t] = __builtin_amdgcn_wmma_f32_16x16x32_bf16(
                false, pf.v, false, vf.v, (short)0, Ot[t], false, false);
        }
    }

    const float sinkv = sinks[h];
    float osc[8];
    #pragma unroll
    for (int i = 0; i < 8; ++i) {
        const float lse = rmax[i] + __logf(rsum[i]);
        const float sig = 1.0f / (1.0f + __expf(-(lse - sinkv)));
        osc[i] = sig / rsum[i];
    }
    #pragma unroll
    for (int t = 0; t < 8; ++t) {
        const int col = h * HD + t * 16 + lo;
        #pragma unroll
        for (int i = 0; i < 8; ++i) {
            const int row = m0 + i + 8 * hf;
            att[(size_t)row * D_MODEL + col] = f2bf(Ot[t][i] * osc[i]);
        }
    }
}

// ---------------------------------------------------------------------------
// Launch
// ---------------------------------------------------------------------------
extern "C" void kernel_launch(void* const* d_in, const int* in_sizes, int n_in,
                              void* d_out, int out_size, void* d_ws, size_t ws_size,
                              hipStream_t stream) {
    const float* x      = (const float*)d_in[0];
    const float* freqs  = (const float*)d_in[1];
    const float* wq     = (const float*)d_in[2];
    const float* wk     = (const float*)d_in[3];
    const float* wv     = (const float*)d_in[4];
    const float* wo     = (const float*)d_in[5];
    const float* sinks  = (const float*)d_in[6];
    float*       out    = (float*)d_out;

    // Workspace layout (bytes) — total ~75.5 MB
    char* w = (char*)d_ws;
    unsigned short* x_bf   = (unsigned short*)(w);                       //  8 MB
    unsigned short* wqkv   = (unsigned short*)(w + 8388608);             // 12 MB
    unsigned short* wo_bf  = (unsigned short*)(w + 20971520);            //  8 MB
    float*          qkv    = (float*)        (w + 29360128);             // 24 MB
    unsigned short* q_bf   = (unsigned short*)(w + 54525952);            //  8 MB
    unsigned short* k_bf   = (unsigned short*)(w + 62914560);            //  2 MB
    unsigned short* vT_bf  = (unsigned short*)(w + 65011712);            //  2 MB
    unsigned short* att_bf = (unsigned short*)(w + 67108864);            //  8 MB

    const int nx = D_MODEL * S_LEN;
    const int nq = NH  * HD * D_MODEL;
    const int nk = NKV * HD * D_MODEL;
    cvt_bf16_kernel<<<(nx + 255) / 256, 256, 0, stream>>>(x,  x_bf, nx);
    cvt_bf16_kernel<<<(nq + 255) / 256, 256, 0, stream>>>(wq, wqkv, nq);
    cvt_bf16_kernel<<<(nk + 255) / 256, 256, 0, stream>>>(wk, wqkv + (size_t)2048 * D_MODEL, nk);
    cvt_bf16_kernel<<<(nk + 255) / 256, 256, 0, stream>>>(wv, wqkv + (size_t)2560 * D_MODEL, nk);
    cvt_bf16_kernel<<<(nq + 255) / 256, 256, 0, stream>>>(wo, wo_bf, nq);

    // qkv = x @ [wq;wk;wv]^T      M=2048 N=3072 K=2048, 128x128 block tiles
    gemm_bf16_kernel<<<(S_LEN / 128) * (QKV_N / 128), 256, 0, stream>>>(
        x_bf, wqkv, qkv, S_LEN, QKV_N, D_MODEL);

    norm_rope_kernel<<<S_LEN * 24, 128, 0, stream>>>(qkv, freqs, q_bf, k_bf, vT_bf);

    {
        dim3 grid(S_LEN / 16, NH);
        flash_attn_kernel<<<grid, 32, 0, stream>>>(q_bf, k_bf, vT_bf, sinks, att_bf);
    }

    // out = att @ wo^T             M=2048 N=2048 K=2048
    gemm_bf16_kernel<<<(S_LEN / 128) * (D_MODEL / 128), 256, 0, stream>>>(
        att_bf, wo_bf, out, S_LEN, D_MODEL, D_MODEL);

    (void)in_sizes; (void)n_in; (void)out_size; (void)ws_size;
}
